// RoPE_MHA_13846974562804
// MI455X (gfx1250) — matrix operand order repaired and checked by reading, not compiled, not run on hardware
//
#include <hip/hip_runtime.h>
#include <hip/hip_bf16.h>

#define B_  2
#define S_  2048
#define D_  1024
#define H_  16
#define HS_ 64

typedef __attribute__((ext_vector_type(16))) __bf16 v16bf;
typedef __attribute__((ext_vector_type(2)))  __bf16 bf16x2;
typedef __attribute__((ext_vector_type(8)))  float  v8f;
typedef int v4i_gcc __attribute__((vector_size(16)));   // payload type the async builtin wants

#if defined(__has_builtin)
#if __has_builtin(__builtin_amdgcn_global_load_async_to_lds_b128)
#define HAVE_ASYNC_LDS 1
#endif
#endif

// ---------------------------------------------------------------------------
// CDNA5 async copy: GLOBAL_LOAD_ASYNC_TO_LDS_B128 (ASYNCcnt) with sync fallback
// ---------------------------------------------------------------------------
__device__ inline void lds_async_b128(const void* g, void* l) {
#if defined(HAVE_ASYNC_LDS)
  // addrspace cast on void* first, then pointee bitcast to the int4 payload type
  __attribute__((address_space(1))) void* g1 =
      (__attribute__((address_space(1))) void*)(uintptr_t)g;       // inttoptr (flat==global)
  __attribute__((address_space(3))) void* l3 =
      (__attribute__((address_space(3))) void*)l;                  // addrspacecast
  __builtin_amdgcn_global_load_async_to_lds_b128(
      (__attribute__((address_space(1))) v4i_gcc*)g1,
      (__attribute__((address_space(3))) v4i_gcc*)l3, 0, 0);
#else
  *(uint4*)l = *(const uint4*)g;
#endif
}

__device__ inline void wait_async() {
#if defined(HAVE_ASYNC_LDS)
#if __has_builtin(__builtin_amdgcn_s_wait_asynccnt)
  __builtin_amdgcn_s_wait_asynccnt(0);
#else
  asm volatile("s_wait_asynccnt 0" ::: "memory");
#endif
#endif
}

// ---------------------------------------------------------------------------
// WMMA helpers (CDNA5 v_wmma_f32_16x16x32_bf16, wave32)
// ---------------------------------------------------------------------------
__device__ inline v8f wmma_bf16(v16bf a, v16bf b, v8f c) {
  // (neg_a, A, neg_b, B, c_mod, C, reuse_a, reuse_b)
  return __builtin_amdgcn_wmma_f32_16x16x32_bf16(false, a, false, b, (short)0, c,
                                                 false, false);
}

// A-matrix 16x32 bf16 fragment. base = row-major 16xK tile start, leading dim
// ldk. ISA layout: lanes 0-15 -> M=lane, VGPR g<4 : K=2g,2g+1 ; g>=4 : K=16+..
//                 lanes 16-31 -> same M, K offset +8 within each half.
__device__ inline v16bf load_afrag(const __bf16* base, int ldk, int koff) {
  const int l  = threadIdx.x & 31;
  const int mm = l & 15;
  const int hh = l >> 4;
  const __bf16* rp = base + (size_t)mm * ldk + koff;
  v16bf a;
#pragma unroll
  for (int g = 0; g < 8; ++g) {
    const int k = (g < 4) ? (hh * 8 + 2 * g) : (16 + hh * 8 + 2 * (g - 4));
    bf16x2 t = *(const bf16x2*)(rp + k);
    a[2 * g]     = t[0];
    a[2 * g + 1] = t[1];
  }
  return a;
}

// B-matrix 32x16 bf16 fragment read from TRANSPOSED storage baseT[n][k]
// (leading dim ldk). Lane n=l%16 selects column, same K layout as A-frag.
__device__ inline v16bf load_bfrag(const __bf16* baseT, int ldk, int koff) {
  const int l  = threadIdx.x & 31;
  const int nn = l & 15;
  const int kh = (l >> 4) << 3;
  const __bf16* rp = baseT + (size_t)nn * ldk + koff + kh;
  v16bf b;
#pragma unroll
  for (int g = 0; g < 8; ++g) {
    bf16x2 t = *(const bf16x2*)(rp + ((g < 4) ? 2 * g : 16 + 2 * (g - 4)));
    b[2 * g]     = t[0];
    b[2 * g + 1] = t[1];
  }
  return b;
}

__device__ inline v8f zero_v8f() {
  v8f z;
#pragma unroll
  for (int i = 0; i < 8; ++i) z[i] = 0.0f;
  return z;
}

// ---------------------------------------------------------------------------
// Kernel 0: fp32 -> bf16 convert
// ---------------------------------------------------------------------------
__global__ void f32_to_bf16_kernel(const float* __restrict__ in,
                                   __bf16* __restrict__ out, int n) {
  int i = blockIdx.x * blockDim.x + threadIdx.x;
  if (i < n) out[i] = (__bf16)in[i];
}

// ---------------------------------------------------------------------------
// Kernel 1/4: C(fp32, MxN) = A(bf16, MxK) @ W(fp32->bf16, KxN) + bias
// 64x64 block tile, 4 waves, each wave owns a 16-row strip (4 WMMA C tiles).
// A tile staged via async-to-LDS; W tile converted fp32->bf16 + transposed.
// ---------------------------------------------------------------------------
__global__ __launch_bounds__(128)
void gemm_bf16_kernel(const __bf16* __restrict__ A, const float* __restrict__ W,
                      const float* __restrict__ bias, float* __restrict__ C,
                      int M, int N, int K) {
  __shared__ __align__(16) __bf16 As[64][32];   // M x K tile
  __shared__ __align__(16) __bf16 WsT[64][32];  // N x K (transposed) tile

  const int tid  = threadIdx.x;
  const int wave = tid >> 5;
  const int lane = tid & 31;
  const int n0   = blockIdx.x * 64;
  const int m0   = blockIdx.y * 64;
  const int mw   = m0 + wave * 16;

  v8f acc[4];
#pragma unroll
  for (int t = 0; t < 4; ++t) acc[t] = zero_v8f();

  for (int k0 = 0; k0 < K; k0 += 32) {
    __syncthreads();
    // ---- A tile: 64x32 bf16 = 256 x 16B, 2 per thread, async copy to LDS
#pragma unroll
    for (int i = 0; i < 2; ++i) {
      int slot = tid * 2 + i;           // 0..255
      int row  = slot >> 2;
      int c8   = (slot & 3) * 8;
      lds_async_b128(A + (size_t)(m0 + row) * K + k0 + c8, &As[row][c8]);
    }
    // ---- W tile: read fp32 rows (float4), store transposed bf16
#pragma unroll
    for (int i = 0; i < 4; ++i) {
      int slot = tid * 4 + i;           // 0..511
      int kk   = slot >> 4;             // 0..31
      int n4   = (slot & 15) * 4;
      const float* wp = W + (size_t)(k0 + kk) * N + n0 + n4;
      float4 wv = *(const float4*)wp;
      WsT[n4 + 0][kk] = (__bf16)wv.x;
      WsT[n4 + 1][kk] = (__bf16)wv.y;
      WsT[n4 + 2][kk] = (__bf16)wv.z;
      WsT[n4 + 3][kk] = (__bf16)wv.w;
    }
    // pull next W tile toward cache while we compute this step
    if (k0 + 32 < K)
      __builtin_prefetch(W + (size_t)(k0 + 32 + (tid >> 2)) * N + n0 + (tid & 3) * 16, 0, 0);
    wait_async();
    __syncthreads();

    // load all fragments first, then stream the WMMA chain (one dscnt wait)
    v16bf af = load_afrag(&As[wave * 16][0], 32, 0);
    v16bf bf[4];
#pragma unroll
    for (int t = 0; t < 4; ++t) bf[t] = load_bfrag(&WsT[t * 16][0], 32, 0);
#pragma unroll
    for (int t = 0; t < 4; ++t) acc[t] = wmma_bf16(af, bf[t], acc[t]);
  }

  // ---- epilogue: C layout N=lane%16, M = r + 8*(lane/16)
  const int nl = lane & 15;
  const int hf = lane >> 4;
#pragma unroll
  for (int t = 0; t < 4; ++t) {
#pragma unroll
    for (int r = 0; r < 8; ++r) {
      int row = mw + r + 8 * hf;
      int col = n0 + t * 16 + nl;
      C[(size_t)row * N + col] = acc[t][r] + bias[col];
    }
  }
}

// ---------------------------------------------------------------------------
// Kernel 2: RoPE prep. Replicates reference exactly:
//   q  = rot(q_proj, fq) ; k = rot(q, fk)  (k built from rotated q!)
//   next_prefix_kv = stack([k_proj, v])    (PRE-rotary k)
// Writes bf16 Q/K/V in (B,H,S,HS) for attention + fp32 prefix output.
// ---------------------------------------------------------------------------
__global__ void rope_prep_kernel(const float* __restrict__ xp,
                                 const float* __restrict__ fq,
                                 const float* __restrict__ fk,
                                 __bf16* __restrict__ Qr,
                                 __bf16* __restrict__ Kr,
                                 __bf16* __restrict__ Vb,
                                 float* __restrict__ prefix_out) {
  int idx = blockIdx.x * blockDim.x + threadIdx.x; // B*H*S*(HS/2) = 2^21
  if (idx >= B_ * H_ * S_ * (HS_ / 2)) return;
  int p = idx & 31;
  int s = (idx >> 5) & (S_ - 1);
  int h = (idx >> 16) & (H_ - 1);
  int b = idx >> 20;

  const float* row = xp + (size_t)(b * S_ + s) * (3 * D_) + h * (3 * HS_);
  float q0 = row[2 * p],           q1 = row[2 * p + 1];
  float k0 = row[HS_ + 2 * p],     k1 = row[HS_ + 2 * p + 1];
  float v0 = row[2 * HS_ + 2 * p], v1 = row[2 * HS_ + 2 * p + 1];

  float fr = fq[((size_t)s * 32 + p) * 2 + 0];
  float fi = fq[((size_t)s * 32 + p) * 2 + 1];
  float qr0 = fr * q0 - fi * q1;
  float qr1 = fr * q1 + fi * q0;

  float gr = fk[((size_t)s * 32 + p) * 2 + 0];
  float gi = fk[((size_t)s * 32 + p) * 2 + 1];
  float kr0 = gr * qr0 - gi * qr1;
  float kr1 = gr * qr1 + gi * qr0;

  size_t base = (((size_t)(b * H_ + h)) * S_ + s) * HS_ + 2 * p;
  Qr[base] = (__bf16)qr0; Qr[base + 1] = (__bf16)qr1;
  Kr[base] = (__bf16)kr0; Kr[base + 1] = (__bf16)kr1;
  Vb[base] = (__bf16)v0;  Vb[base + 1] = (__bf16)v1;

  const size_t kvplane = (size_t)B_ * H_ * S_ * HS_;
  prefix_out[base]               = k0;
  prefix_out[base + 1]           = k1;
  prefix_out[kvplane + base]     = v0;
  prefix_out[kvplane + base + 1] = v1;
}

// ---------------------------------------------------------------------------
// Kernel 3: flash attention (bf16 WMMA, f32 accum, online softmax).
// Block = 4 waves x 16 q-rows = 64 q-rows; loop over S in 32-key chunks.
// K chunk staged via async-to-LDS; V chunk transposed manually.
// ---------------------------------------------------------------------------
__global__ __launch_bounds__(128)
void flash_attn_kernel(const __bf16* __restrict__ Qr,
                       const __bf16* __restrict__ Kr,
                       const __bf16* __restrict__ Vb,
                       const int* __restrict__ mask,
                       __bf16* __restrict__ AO) {
  __shared__ __align__(16) __bf16 Ks[32][64];      // keyrow x hs
  __shared__ __align__(16) __bf16 Vt[64][32];      // hs x keyrow (transposed)
  __shared__ __align__(16) __bf16 Pb[4][16][32];   // per-wave P scratch

  const int tid  = threadIdx.x;
  const int wave = tid >> 5;
  const int lane = tid & 31;
  const int bid  = blockIdx.x;
  const int qt   = bid & (S_ / 64 - 1);
  const int h    = (bid >> 5) & (H_ - 1);
  const int b    = bid >> 9;

  const size_t headbase = ((size_t)(b * H_ + h)) * S_ * HS_;
  const int qbase = qt * 64 + wave * 16;   // sequence row of this wave's tile

  // Q fragments (16x64 per wave -> two K=32 A-frags), loaded once from global
  const __bf16* qtile = Qr + headbase + (size_t)qbase * HS_;
  v16bf qa0 = load_afrag(qtile, HS_, 0);
  v16bf qa1 = load_afrag(qtile, HS_, 32);

  float mrow[8], lrow[8];
  v8f o[4];
#pragma unroll
  for (int r = 0; r < 8; ++r) { mrow[r] = -3.0e38f; lrow[r] = 0.0f; }
#pragma unroll
  for (int t = 0; t < 4; ++t) o[t] = zero_v8f();

  const int nl = lane & 15;
  const int hf = lane >> 4;
  const float scale = 0.125f;  // 1/sqrt(64)

  for (int s0 = 0; s0 < S_; s0 += 32) {
    __syncthreads();
    // ---- stage K chunk via async copy, V chunk transposed into LDS
#pragma unroll
    for (int i = 0; i < 2; ++i) {
      int slot = tid * 2 + i;            // 0..255
      int kr   = slot >> 3;              // keyrow 0..31
      int c8   = (slot & 7) * 8;         // hs offset
      lds_async_b128(Kr + headbase + (size_t)(s0 + kr) * HS_ + c8, &Ks[kr][c8]);
    }
#pragma unroll
    for (int i = 0; i < 2; ++i) {
      int slot = tid * 2 + i;
      int kr   = slot >> 3;
      int c8   = (slot & 7) * 8;
      uint4 vv = *(const uint4*)(Vb + headbase + (size_t)(s0 + kr) * HS_ + c8);
      const __bf16* pv = (const __bf16*)&vv;
#pragma unroll
      for (int j = 0; j < 8; ++j) Vt[c8 + j][kr] = pv[j];
    }
    wait_async();
    __syncthreads();

    // ---- scores: 16x32 = two 16x16 C tiles, each sum of two K=32 WMMAs
    v16bf kb[2][2];
#pragma unroll
    for (int t = 0; t < 2; ++t) {
      kb[t][0] = load_bfrag(&Ks[t * 16][0], 64, 0);
      kb[t][1] = load_bfrag(&Ks[t * 16][0], 64, 32);
    }
    v8f sc[2];
#pragma unroll
    for (int t = 0; t < 2; ++t) {
      sc[t] = wmma_bf16(qa0, kb[t][0], zero_v8f());
      sc[t] = wmma_bf16(qa1, kb[t][1], sc[t]);
    }

    // ---- mask + scale
    float pv[2][8];
#pragma unroll
    for (int t = 0; t < 2; ++t)
#pragma unroll
      for (int r = 0; r < 8; ++r) {
        int qrow = qbase + r + 8 * hf;
        int kcol = s0 + t * 16 + nl;
        int mv = mask[(size_t)qrow * S_ + kcol];
        pv[t][r] = (mv == 0) ? -65504.0f : sc[t][r] * scale;
      }

    // ---- online softmax (row spread over 16 lanes of a half -> shfl_xor)
#pragma unroll
    for (int r = 0; r < 8; ++r) {
      float mx = fmaxf(pv[0][r], pv[1][r]);
#pragma unroll
      for (int d = 1; d < 16; d <<= 1) mx = fmaxf(mx, __shfl_xor(mx, d, 32));
      float mnew = fmaxf(mrow[r], mx);
      float corr = __expf(mrow[r] - mnew);
      float p0 = __expf(pv[0][r] - mnew);
      float p1 = __expf(pv[1][r] - mnew);
      pv[0][r] = p0; pv[1][r] = p1;
      float rs = p0 + p1;
#pragma unroll
      for (int d = 1; d < 16; d <<= 1) rs += __shfl_xor(rs, d, 32);
      lrow[r] = lrow[r] * corr + rs;
      mrow[r] = mnew;
#pragma unroll
      for (int t = 0; t < 4; ++t) o[t][r] *= corr;
    }

    // ---- P: C layout -> LDS -> A layout
#pragma unroll
    for (int t = 0; t < 2; ++t)
#pragma unroll
      for (int r = 0; r < 8; ++r)
        Pb[wave][r + 8 * hf][t * 16 + nl] = (__bf16)pv[t][r];
    __syncthreads();

    v16bf pa = load_afrag(&Pb[wave][0][0], 32, 0);
    v16bf vb[4];
#pragma unroll
    for (int t = 0; t < 4; ++t) vb[t] = load_bfrag(&Vt[t * 16][0], 32, 0);
#pragma unroll
    for (int t = 0; t < 4; ++t) o[t] = wmma_bf16(pa, vb[t], o[t]);
  }

  // ---- normalize and write attn_o (B,S,D) as bf16
#pragma unroll
  for (int r = 0; r < 8; ++r) {
    float inv = 1.0f / lrow[r];
    int qrow = qbase + r + 8 * hf;
#pragma unroll
    for (int t = 0; t < 4; ++t) {
      int hs = t * 16 + nl;
      AO[((size_t)(b * S_ + qrow)) * D_ + h * HS_ + hs] = (__bf16)(o[t][r] * inv);
    }
  }
}

// ---------------------------------------------------------------------------
// Launch: x->bf16 | QKV GEMM | RoPE prep (+prefix out) | flash attn | FF GEMM
// ---------------------------------------------------------------------------
extern "C" void kernel_launch(void* const* d_in, const int* in_sizes, int n_in,
                              void* d_out, int out_size, void* d_ws, size_t ws_size,
                              hipStream_t stream) {
  (void)in_sizes; (void)n_in; (void)out_size; (void)ws_size;

  const float* x      = (const float*)d_in[0];
  const int*   mask   = (const int*)d_in[1];
  const float* fq     = (const float*)d_in[2];
  const float* fk     = (const float*)d_in[3];
  /* d_in[4] = prefix_kv, zero-length, unused */
  const float* w_proj = (const float*)d_in[5];
  const float* b_proj = (const float*)d_in[6];
  const float* w_ff   = (const float*)d_in[7];
  const float* b_ff   = (const float*)d_in[8];

  float* out        = (float*)d_out;
  float* prefix_out = out + (size_t)B_ * S_ * D_;

  // workspace carve-up
  char* ws = (char*)d_ws;
  __bf16* Xb = (__bf16*)ws; ws += (size_t)B_ * S_ * D_ * 2;        // x in bf16
  float*  xp = (float*)ws;  ws += (size_t)B_ * S_ * 3 * D_ * 4;    // qkv proj
  __bf16* Qr = (__bf16*)ws; ws += (size_t)B_ * S_ * D_ * 2;
  __bf16* Kr = (__bf16*)ws; ws += (size_t)B_ * S_ * D_ * 2;
  __bf16* Vb = (__bf16*)ws; ws += (size_t)B_ * S_ * D_ * 2;
  __bf16* AO = (__bf16*)ws; ws += (size_t)B_ * S_ * D_ * 2;

  const int nX = B_ * S_ * D_;
  f32_to_bf16_kernel<<<(nX + 255) / 256, 256, 0, stream>>>(x, Xb, nX);

  gemm_bf16_kernel<<<dim3(3 * D_ / 64, (B_ * S_) / 64), 128, 0, stream>>>(
      Xb, w_proj, b_proj, xp, B_ * S_, 3 * D_, D_);

  const int nP = B_ * H_ * S_ * (HS_ / 2);
  rope_prep_kernel<<<(nP + 255) / 256, 256, 0, stream>>>(xp, fq, fk, Qr, Kr, Vb,
                                                         prefix_out);

  flash_attn_kernel<<<B_ * H_ * (S_ / 64), 128, 0, stream>>>(Qr, Kr, Vb, mask, AO);

  gemm_bf16_kernel<<<dim3(D_ / 64, (B_ * S_) / 64), 128, 0, stream>>>(
      AO, w_ff, b_ff, out, B_ * S_, D_, D_);
}
